// Qwen3MoeDecoderLayer_55559696941216
// MI455X (gfx1250) — compile-verified, matrix operand order checked by reading, not run on hardware
//
#include <hip/hip_runtime.h>

// ---------------------------------------------------------------------------
// Qwen3-MoE decoder layer for gfx1250 (MI455X), wave32 + WMMA f32_16x16x32_f16
// Double-buffered LDS staging; f16 side buffers for attention operands.
// ---------------------------------------------------------------------------

typedef __attribute__((ext_vector_type(16))) _Float16 v16h;
typedef __attribute__((ext_vector_type(8)))  _Float16 v8h;
typedef __attribute__((ext_vector_type(2)))  _Float16 v2h;
typedef __attribute__((ext_vector_type(8)))  float    v8f;

#define LDS_STRIDE 40   // 32 K-halves padded to 40 (80B row: 16B-aligned, conflict-free)

static constexpr int Sq  = 1024;
static constexpr int Hd  = 2048;
static constexpr int NHq = 32;
static constexpr int NKV = 4;
static constexpr int HDq = 128;
static constexpr int NE  = 32;
static constexpr int TOPK = 8;
static constexpr int NI  = 768;

// ---- WMMA fragment loader -------------------------------------------------
// 16-bit A/B matrix layout (ISA 7.12.2): lane l -> row l&15; lanes 0-15 carry
// K 0..7 / 16..23, lanes 16-31 carry K 8..15 / 24..31 => two b128 LDS reads.
__device__ inline v16h load_frag(const _Float16* __restrict__ tile, int lane) {
    int r    = lane & 15;
    int half = lane >> 4;
    const _Float16* p = tile + r * LDS_STRIDE + half * 8;
    v8h lo = *(const v8h*)p;
    v8h hi = *(const v8h*)(p + 16);
    v16h out;
    *(v8h*)&out       = lo;
    *((v8h*)&out + 1) = hi;
    return out;
}

// Packed 2xK staging store: LDS layout [n][k], consecutive k pair as one b32.
template<typename T>
__device__ inline void store_pair(_Float16* __restrict__ lds, int n, int kpair,
                                  T lo, T hi) {
    v2h t;
    t[0] = (_Float16)lo;
    t[1] = (_Float16)hi;
    *(v2h*)&lds[n * LDS_STRIDE + kpair] = t;
}

// ---------------------------------------------------------------------------
// 64x64 GEMM (attention-sized), z-batched, double-buffered.
// TA/TB = float (convert on stage) or _Float16 (pure copy).
// B row-major KxN, or NxK if TRANSB.
// ---------------------------------------------------------------------------
template<typename TA, typename TB, bool TRANSB, bool ADD_R>
__launch_bounds__(256)
__global__ void gemm_tile64(const TA* __restrict__ A, const TB* __restrict__ B,
                            const float* __restrict__ R, float* __restrict__ C,
                            int M, int N, int K, int lda, int ldb, int ldc,
                            long long aZ, long long bZ, long long cZ)
{
    A += (size_t)blockIdx.z * aZ;
    B += (size_t)blockIdx.z * bZ;
    C += (size_t)blockIdx.z * cZ;
    if (ADD_R) R += (size_t)blockIdx.z * cZ;

    __shared__ _Float16 Alds[2][64 * LDS_STRIDE];
    __shared__ _Float16 Blds[2][64 * LDS_STRIDE];
    const int tid  = threadIdx.x;
    const int n0   = blockIdx.x * 64;
    const int m0   = blockIdx.y * 64;
    const int lane = tid & 31;
    const int wave = tid >> 5;
    const int wm   = wave & 3;   // 4 waves x 16 rows
    const int wn   = wave >> 2;  // 2 waves x 32 cols

    v8f acc0 = {};
    v8f acc1 = {};

    auto stage = [&](int k0, _Float16* __restrict__ As, _Float16* __restrict__ Bs) {
        {   // A tile 64x32
            int row = tid >> 2;
            int c0  = (tid & 3) * 8;
            v8h t;
            if (m0 + row < M) {
                const TA* g = A + (size_t)(m0 + row) * lda + k0 + c0;
#pragma unroll
                for (int i = 0; i < 8; ++i) t[i] = (_Float16)g[i];
            } else {
#pragma unroll
                for (int i = 0; i < 8; ++i) t[i] = (_Float16)0.f;
            }
            *(v8h*)&As[row * LDS_STRIDE + c0] = t;
        }
        if constexpr (TRANSB) {   // B is NxK: contiguous k, same as A staging
            int row = tid >> 2;
            int c0  = (tid & 3) * 8;
            const TB* g = B + (size_t)(n0 + row) * ldb + k0 + c0;
            v8h t;
#pragma unroll
            for (int i = 0; i < 8; ++i) t[i] = (_Float16)g[i];
            *(v8h*)&Bs[row * LDS_STRIDE + c0] = t;
        } else {                  // B is KxN: pack k-pairs, one b32 store per n
            int kp  = (tid >> 4) * 2;       // 0..30
            int nc0 = (tid & 15) * 4;       // 0..60
            const TB* g0 = B + (size_t)(k0 + kp) * ldb + n0 + nc0;
            const TB* g1 = g0 + ldb;
#pragma unroll
            for (int i = 0; i < 4; ++i) store_pair(Bs, nc0 + i, kp, g0[i], g1[i]);
        }
    };

    stage(0, Alds[0], Blds[0]);
    const int nk = K >> 5;
    int p = 0;
    for (int kt = 0; kt < nk; ++kt) {
        __syncthreads();
        if (kt + 1 < nk) stage((kt + 1) << 5, Alds[p ^ 1], Blds[p ^ 1]);
        v16h a  = load_frag(&Alds[p][(wm * 16) * LDS_STRIDE], lane);
        v16h b0 = load_frag(&Blds[p][(wn * 32) * LDS_STRIDE], lane);
        v16h b1 = load_frag(&Blds[p][(wn * 32 + 16) * LDS_STRIDE], lane);
        acc0 = __builtin_amdgcn_wmma_f32_16x16x32_f16(false, a, false, b0, (short)0, acc0, false, false);
        acc1 = __builtin_amdgcn_wmma_f32_16x16x32_f16(false, a, false, b1, (short)0, acc1, false, false);
        p ^= 1;
    }

    const int half = lane >> 4;
    const int nc   = lane & 15;
#pragma unroll
    for (int i = 0; i < 8; ++i) {
        int m = m0 + wm * 16 + half * 8 + i;
        if (m < M) {
            int n = n0 + wn * 32 + nc;
            float v = acc0[i];
            if (ADD_R) v += R[(size_t)m * ldc + n];
            C[(size_t)m * ldc + n] = v;
            n += 16;
            v = acc1[i];
            if (ADD_R) v += R[(size_t)m * ldc + n];
            C[(size_t)m * ldc + n] = v;
        }
    }
}

// ---------------------------------------------------------------------------
// 128x128 GEMM for weight-bound projections (fp32 weights, convert on stage),
// double-buffered. 8 WMMAs per wave per K-step.
// ---------------------------------------------------------------------------
template<bool ADD_R>
__launch_bounds__(256)
__global__ void gemm128_f16w(const float* __restrict__ A, const float* __restrict__ B,
                             const float* __restrict__ R, float* __restrict__ C,
                             int M, int N, int K, int lda, int ldb, int ldc)
{
    __shared__ _Float16 Alds[2][128 * LDS_STRIDE];
    __shared__ _Float16 Blds[2][128 * LDS_STRIDE];
    const int tid  = threadIdx.x;
    const int n0   = blockIdx.x * 128;
    const int m0   = blockIdx.y * 128;
    const int lane = tid & 31;
    const int wave = tid >> 5;
    const int wm   = wave & 3;   // 4 waves x 32 rows (2 m-frags)
    const int wn   = wave >> 2;  // 2 waves x 64 cols (4 n-frags)

    v8f acc[2][4] = {};

    auto stage = [&](int k0, _Float16* __restrict__ As, _Float16* __restrict__ Bs) {
        {   // A tile 128x32: 2 threads per row, 16 floats each
            int row = tid >> 1;
            int c0  = (tid & 1) * 16;
            v8h t0, t1;
            if (m0 + row < M) {
                const float* g = A + (size_t)(m0 + row) * lda + k0 + c0;
#pragma unroll
                for (int i = 0; i < 8; ++i) t0[i] = (_Float16)g[i];
#pragma unroll
                for (int i = 0; i < 8; ++i) t1[i] = (_Float16)g[8 + i];
            } else {
#pragma unroll
                for (int i = 0; i < 8; ++i) { t0[i] = (_Float16)0.f; t1[i] = (_Float16)0.f; }
            }
            *(v8h*)&As[row * LDS_STRIDE + c0]     = t0;
            *(v8h*)&As[row * LDS_STRIDE + c0 + 8] = t1;
        }
        {   // B tile 32x128: k-pairs packed, 8 n per thread
            int kp  = (tid >> 4) * 2;     // 0..30
            int nc0 = (tid & 15) * 8;     // 0..120
            const float* g0 = B + (size_t)(k0 + kp) * ldb + n0 + nc0;
            const float* g1 = g0 + ldb;
#pragma unroll
            for (int i = 0; i < 8; ++i) store_pair(Bs, nc0 + i, kp, g0[i], g1[i]);
        }
    };

    stage(0, Alds[0], Blds[0]);
    const int nk = K >> 5;
    int p = 0;
    for (int kt = 0; kt < nk; ++kt) {
        __syncthreads();
        if (kt + 1 < nk) stage((kt + 1) << 5, Alds[p ^ 1], Blds[p ^ 1]);
        v16h a0 = load_frag(&Alds[p][(wm * 32) * LDS_STRIDE], lane);
        v16h a1 = load_frag(&Alds[p][(wm * 32 + 16) * LDS_STRIDE], lane);
#pragma unroll
        for (int jn = 0; jn < 4; ++jn) {
            v16h b = load_frag(&Blds[p][(wn * 64 + jn * 16) * LDS_STRIDE], lane);
            acc[0][jn] = __builtin_amdgcn_wmma_f32_16x16x32_f16(false, a0, false, b, (short)0, acc[0][jn], false, false);
            acc[1][jn] = __builtin_amdgcn_wmma_f32_16x16x32_f16(false, a1, false, b, (short)0, acc[1][jn], false, false);
        }
        p ^= 1;
    }

    const int half = lane >> 4;
    const int nc   = lane & 15;
#pragma unroll
    for (int jm = 0; jm < 2; ++jm) {
#pragma unroll
        for (int i = 0; i < 8; ++i) {
            int m = m0 + wm * 32 + jm * 16 + half * 8 + i;
            if (m < M) {
#pragma unroll
                for (int jn = 0; jn < 4; ++jn) {
                    int n = n0 + wn * 64 + jn * 16 + nc;
                    float v = acc[jm][jn][i];
                    if (ADD_R) v += R[(size_t)m * ldc + n];
                    C[(size_t)m * ldc + n] = v;
                }
            }
        }
    }
}

// ---------------------------------------------------------------------------
// MoE up-projection, fused gate&up&SiLU, 128x64 tiles, gathered A rows,
// double-buffered.
// ---------------------------------------------------------------------------
__launch_bounds__(256)
__global__ void moe_up_kernel(const float* __restrict__ H2,
                              const float* __restrict__ WG, const float* __restrict__ WU,
                              float* __restrict__ ACT,
                              const int* __restrict__ cnt, const int* __restrict__ off,
                              const int* __restrict__ entryTok)
{
    const int e  = blockIdx.z;
    const int ce = cnt[e];
    const int m0 = blockIdx.y * 128;
    if (m0 >= ce) return;
    const int n0   = blockIdx.x * 64;
    const int base = off[e];
    const float* BG = WG + (size_t)e * Hd * NI;
    const float* BU = WU + (size_t)e * Hd * NI;

    __shared__ _Float16 Alds[2][128 * LDS_STRIDE];
    __shared__ _Float16 Glds[2][64 * LDS_STRIDE];
    __shared__ _Float16 Ulds[2][64 * LDS_STRIDE];
    const int tid  = threadIdx.x;
    const int lane = tid & 31;
    const int wave = tid >> 5;
    const int wm   = wave & 3;   // 4 waves x 32 rows
    const int wn   = wave >> 2;  // 2 waves x 32 cols

    v8f ag[2][2] = {};
    v8f au[2][2] = {};

    auto stage = [&](int k0, _Float16* __restrict__ As,
                     _Float16* __restrict__ Gs, _Float16* __restrict__ Us) {
        {   // gathered A tile 128x32
            int row = tid >> 1;
            int c0  = (tid & 1) * 16;
            v8h t0, t1;
            if (m0 + row < ce) {
                int tok = entryTok[base + m0 + row];
                const float* g = H2 + (size_t)tok * Hd + k0 + c0;
#pragma unroll
                for (int i = 0; i < 8; ++i) t0[i] = (_Float16)g[i];
#pragma unroll
                for (int i = 0; i < 8; ++i) t1[i] = (_Float16)g[8 + i];
            } else {
#pragma unroll
                for (int i = 0; i < 8; ++i) { t0[i] = (_Float16)0.f; t1[i] = (_Float16)0.f; }
            }
            *(v8h*)&As[row * LDS_STRIDE + c0]     = t0;
            *(v8h*)&As[row * LDS_STRIDE + c0 + 8] = t1;
        }
        {   // wg & wu tiles 32x64, packed k-pairs
            int kp  = (tid >> 4) * 2;
            int nc0 = (tid & 15) * 4;
            const float* gg0 = BG + (size_t)(k0 + kp) * NI + n0 + nc0;
            const float* gg1 = gg0 + NI;
            const float* gu0 = BU + (size_t)(k0 + kp) * NI + n0 + nc0;
            const float* gu1 = gu0 + NI;
#pragma unroll
            for (int i = 0; i < 4; ++i) {
                store_pair(Gs, nc0 + i, kp, gg0[i], gg1[i]);
                store_pair(Us, nc0 + i, kp, gu0[i], gu1[i]);
            }
        }
    };

    stage(0, Alds[0], Glds[0], Ulds[0]);
    const int nk = Hd >> 5;
    int p = 0;
    for (int kt = 0; kt < nk; ++kt) {
        __syncthreads();
        if (kt + 1 < nk) stage((kt + 1) << 5, Alds[p ^ 1], Glds[p ^ 1], Ulds[p ^ 1]);
        v16h a0 = load_frag(&Alds[p][(wm * 32) * LDS_STRIDE], lane);
        v16h a1 = load_frag(&Alds[p][(wm * 32 + 16) * LDS_STRIDE], lane);
#pragma unroll
        for (int jn = 0; jn < 2; ++jn) {
            v16h bg = load_frag(&Glds[p][(wn * 32 + jn * 16) * LDS_STRIDE], lane);
            v16h bu = load_frag(&Ulds[p][(wn * 32 + jn * 16) * LDS_STRIDE], lane);
            ag[0][jn] = __builtin_amdgcn_wmma_f32_16x16x32_f16(false, a0, false, bg, (short)0, ag[0][jn], false, false);
            ag[1][jn] = __builtin_amdgcn_wmma_f32_16x16x32_f16(false, a1, false, bg, (short)0, ag[1][jn], false, false);
            au[0][jn] = __builtin_amdgcn_wmma_f32_16x16x32_f16(false, a0, false, bu, (short)0, au[0][jn], false, false);
            au[1][jn] = __builtin_amdgcn_wmma_f32_16x16x32_f16(false, a1, false, bu, (short)0, au[1][jn], false, false);
        }
        p ^= 1;
    }

    const int half = lane >> 4;
    const int nc   = lane & 15;
#pragma unroll
    for (int jm = 0; jm < 2; ++jm) {
#pragma unroll
        for (int i = 0; i < 8; ++i) {
            int m = m0 + wm * 32 + jm * 16 + half * 8 + i;
            if (m < ce) {
                size_t row = (size_t)(base + m) * NI;
#pragma unroll
                for (int jn = 0; jn < 2; ++jn) {
                    int n = n0 + wn * 32 + jn * 16 + nc;
                    float g = ag[jm][jn][i];
                    float u = au[jm][jn][i];
                    ACT[row + n] = g * __builtin_amdgcn_rcpf(1.f + __expf(-g)) * u;
                }
            }
        }
    }
}

// ---------------------------------------------------------------------------
// MoE down-projection 128x128 with weighted atomic scatter, double-buffered.
// ---------------------------------------------------------------------------
__launch_bounds__(256)
__global__ void moe_down_kernel(const float* __restrict__ ACT, const float* __restrict__ WD,
                                float* __restrict__ OUT,
                                const int* __restrict__ cnt, const int* __restrict__ off,
                                const int* __restrict__ entryTok, const float* __restrict__ entryW)
{
    const int e  = blockIdx.z;
    const int ce = cnt[e];
    const int m0 = blockIdx.y * 128;
    if (m0 >= ce) return;
    const int n0   = blockIdx.x * 128;
    const int base = off[e];
    const float* B = WD + (size_t)e * NI * Hd;

    __shared__ _Float16 Alds[2][128 * LDS_STRIDE];
    __shared__ _Float16 Blds[2][128 * LDS_STRIDE];
    const int tid  = threadIdx.x;
    const int lane = tid & 31;
    const int wave = tid >> 5;
    const int wm   = wave & 3;
    const int wn   = wave >> 2;

    v8f acc[2][4] = {};

    auto stage = [&](int k0, _Float16* __restrict__ As, _Float16* __restrict__ Bs) {
        {
            int row = tid >> 1;
            int c0  = (tid & 1) * 16;
            v8h t0, t1;
            if (m0 + row < ce) {
                const float* g = ACT + (size_t)(base + m0 + row) * NI + k0 + c0;
#pragma unroll
                for (int i = 0; i < 8; ++i) t0[i] = (_Float16)g[i];
#pragma unroll
                for (int i = 0; i < 8; ++i) t1[i] = (_Float16)g[8 + i];
            } else {
#pragma unroll
                for (int i = 0; i < 8; ++i) { t0[i] = (_Float16)0.f; t1[i] = (_Float16)0.f; }
            }
            *(v8h*)&As[row * LDS_STRIDE + c0]     = t0;
            *(v8h*)&As[row * LDS_STRIDE + c0 + 8] = t1;
        }
        {
            int kp  = (tid >> 4) * 2;
            int nc0 = (tid & 15) * 8;
            const float* g0 = B + (size_t)(k0 + kp) * Hd + n0 + nc0;
            const float* g1 = g0 + Hd;
#pragma unroll
            for (int i = 0; i < 8; ++i) store_pair(Bs, nc0 + i, kp, g0[i], g1[i]);
        }
    };

    stage(0, Alds[0], Blds[0]);
    const int nk = NI >> 5;
    int p = 0;
    for (int kt = 0; kt < nk; ++kt) {
        __syncthreads();
        if (kt + 1 < nk) stage((kt + 1) << 5, Alds[p ^ 1], Blds[p ^ 1]);
        v16h a0 = load_frag(&Alds[p][(wm * 32) * LDS_STRIDE], lane);
        v16h a1 = load_frag(&Alds[p][(wm * 32 + 16) * LDS_STRIDE], lane);
#pragma unroll
        for (int jn = 0; jn < 4; ++jn) {
            v16h b = load_frag(&Blds[p][(wn * 64 + jn * 16) * LDS_STRIDE], lane);
            acc[0][jn] = __builtin_amdgcn_wmma_f32_16x16x32_f16(false, a0, false, b, (short)0, acc[0][jn], false, false);
            acc[1][jn] = __builtin_amdgcn_wmma_f32_16x16x32_f16(false, a1, false, b, (short)0, acc[1][jn], false, false);
        }
        p ^= 1;
    }

    const int half = lane >> 4;
    const int nc   = lane & 15;
#pragma unroll
    for (int jm = 0; jm < 2; ++jm) {
#pragma unroll
        for (int i = 0; i < 8; ++i) {
            int m = m0 + wm * 32 + jm * 16 + half * 8 + i;
            if (m < ce) {
                int   tok = entryTok[base + m];
                float w   = entryW[base + m];
#pragma unroll
                for (int jn = 0; jn < 4; ++jn) {
                    int n = n0 + wn * 64 + jn * 16 + nc;
                    atomicAdd(&OUT[(size_t)tok * Hd + n], w * acc[jm][jn][i]);
                }
            }
        }
    }
}

// ---------------------------------------------------------------------------
// RMSNorm over H=2048 (one block per token, 256 threads x 8 elems)
// ---------------------------------------------------------------------------
__launch_bounds__(256)
__global__ void rmsnorm_kernel(const float* __restrict__ X, const float* __restrict__ W,
                               float* __restrict__ Y)
{
    const int t   = blockIdx.x;
    const int tid = threadIdx.x;
    const float* x = X + (size_t)t * Hd;
    float v[8];
    float ss = 0.f;
#pragma unroll
    for (int j = 0; j < 8; ++j) { v[j] = x[tid + j * 256]; ss += v[j] * v[j]; }
    __shared__ float red[256];
    red[tid] = ss; __syncthreads();
    for (int s = 128; s; s >>= 1) { if (tid < s) red[tid] += red[tid + s]; __syncthreads(); }
    const float r = rsqrtf(red[0] / (float)Hd + 1e-6f);
    float* y = Y + (size_t)t * Hd;
#pragma unroll
    for (int j = 0; j < 8; ++j) { int i = tid + j * 256; y[i] = v[j] * r * W[i]; }
}

// ---------------------------------------------------------------------------
// Per-head RMSNorm (HD=128) + RoPE; reads fp32, writes f16 side buffer.
// grid=(S, nheads), block=64.
// ---------------------------------------------------------------------------
__launch_bounds__(64)
__global__ void qknorm_rope_kernel(const float* __restrict__ inBuf, _Float16* __restrict__ outBuf,
                                   const float* __restrict__ nw, const float* __restrict__ pe,
                                   int nheads)
{
    const int t = blockIdx.x, h = blockIdx.y, d = threadIdx.x;
    const float* p = inBuf + ((size_t)t * nheads + h) * HDq;
    _Float16*   po = outBuf + ((size_t)t * nheads + h) * HDq;
    float x0 = p[2 * d], x1 = p[2 * d + 1];
    __shared__ float red[64];
    red[d] = x0 * x0 + x1 * x1; __syncthreads();
    for (int s = 32; s; s >>= 1) { if (d < s) red[d] += red[d + s]; __syncthreads(); }
    const float r  = rsqrtf(red[0] / (float)HDq + 1e-6f);
    const float n0 = x0 * r * nw[2 * d];
    const float n1 = x1 * r * nw[2 * d + 1];
    const float ang = pe[t * (HDq / 2) + d];
    const float c = __cosf(ang), s = __sinf(ang);
    v2h o;
    o[0] = (_Float16)(n0 * c - n1 * s);
    o[1] = (_Float16)(n0 * s + n1 * c);
    *(v2h*)&po[2 * d] = o;
}

// f32 -> f16 cast (4 elems/thread)
__launch_bounds__(256)
__global__ void cast_f16_kernel(const float* __restrict__ X, _Float16* __restrict__ Y, int n)
{
    int i = (blockIdx.x * 256 + threadIdx.x) * 4;
    if (i < n) {
#pragma unroll
        for (int j = 0; j < 4; ++j) Y[i + j] = (_Float16)X[i + j];
    }
}

// ---------------------------------------------------------------------------
// Causal softmax with 1/sqrt(HD) scale; fp32 scores in, f16 probabilities out.
// grid=(S, nHeadsBatch).
// ---------------------------------------------------------------------------
__launch_bounds__(256)
__global__ void attn_softmax_kernel(const float* __restrict__ scores, _Float16* __restrict__ probs)
{
    const int q   = blockIdx.x;
    const int tid = threadIdx.x;
    const float* row = scores + (size_t)blockIdx.y * Sq * Sq + (size_t)q * Sq;
    _Float16*   prow = probs  + (size_t)blockIdx.y * Sq * Sq + (size_t)q * Sq;
    const float scale = 0.08838834764831845f;  // 128^-0.5
    __shared__ float red[256];
    float mx = -1e30f;
#pragma unroll
    for (int j = 0; j < 4; ++j) {
        int k = tid + j * 256;
        float v = row[k] * scale + (k <= q ? 0.f : -1e9f);
        mx = fmaxf(mx, v);
    }
    red[tid] = mx; __syncthreads();
    for (int s = 128; s; s >>= 1) { if (tid < s) red[tid] = fmaxf(red[tid], red[tid + s]); __syncthreads(); }
    mx = red[0]; __syncthreads();
    float vals[4];
    float sum = 0.f;
#pragma unroll
    for (int j = 0; j < 4; ++j) {
        int k = tid + j * 256;
        float v = row[k] * scale + (k <= q ? 0.f : -1e9f);
        v = __expf(v - mx);
        vals[j] = v; sum += v;
    }
    red[tid] = sum; __syncthreads();
    for (int s = 128; s; s >>= 1) { if (tid < s) red[tid] += red[tid + s]; __syncthreads(); }
    const float inv = 1.f / red[0];
#pragma unroll
    for (int j = 0; j < 4; ++j) prow[tid + j * 256] = (_Float16)(vals[j] * inv);
}

// ---------------------------------------------------------------------------
// Router: one wave per token. logits -> softmax -> top-8 (renormalized).
// ---------------------------------------------------------------------------
__launch_bounds__(32)
__global__ void gate_topk_kernel(const float* __restrict__ H2, const float* __restrict__ GW,
                                 float* __restrict__ topw, int* __restrict__ topi)
{
    const int t    = blockIdx.x;
    const int lane = threadIdx.x;
    const float* h = H2 + (size_t)t * Hd;
    float logit = 0.f;
    for (int i = 0; i < Hd; ++i) logit += h[i] * GW[i * NE + lane];

    float mx = logit;
    for (int o = 16; o; o >>= 1) mx = fmaxf(mx, __shfl_xor(mx, o, 32));
    float p = __expf(logit - mx);
    float sm = p;
    for (int o = 16; o; o >>= 1) sm += __shfl_xor(sm, o, 32);
    p /= sm;

    float selv[TOPK]; int seli[TOPK];
    float pw = p, wsum = 0.f;
#pragma unroll
    for (int s = 0; s < TOPK; ++s) {
        float v = pw; int idx = lane;
        for (int o = 16; o; o >>= 1) {
            float ov = __shfl_xor(v, o, 32);
            int   oi = __shfl_xor(idx, o, 32);
            if (ov > v || (ov == v && oi < idx)) { v = ov; idx = oi; }
        }
        selv[s] = v; seli[s] = idx; wsum += v;
        if (lane == idx) pw = -1.f;
    }
    if (lane == 0) {
        const float inv = 1.f / wsum;
#pragma unroll
        for (int s = 0; s < TOPK; ++s) {
            topw[t * TOPK + s] = selv[s] * inv;
            topi[t * TOPK + s] = seli[s];
        }
    }
}

// ---- routing plumbing ------------------------------------------------------
__global__ void zero_counts_kernel(int* cnt, int* cnt2)
{
    int i = threadIdx.x;
    if (i < NE) { cnt[i] = 0; cnt2[i] = 0; }
}
__global__ void count_kernel(const int* __restrict__ topi, int* __restrict__ cnt)
{
    int g = blockIdx.x * blockDim.x + threadIdx.x;
    if (g < Sq * TOPK) atomicAdd(&cnt[topi[g]], 1);
}
__global__ void scan_kernel(const int* __restrict__ cnt, int* __restrict__ off)
{
    if (threadIdx.x == 0) {
        int a = 0;
        for (int e = 0; e < NE; ++e) { off[e] = a; a += cnt[e]; }
    }
}
__global__ void scatter_kernel(const int* __restrict__ topi, const float* __restrict__ topw,
                               const int* __restrict__ off, int* __restrict__ cnt2,
                               int* __restrict__ entryTok, float* __restrict__ entryW)
{
    int g = blockIdx.x * blockDim.x + threadIdx.x;
    if (g < Sq * TOPK) {
        int t = g >> 3;
        int e = topi[g];
        int pos = atomicAdd(&cnt2[e], 1);
        entryTok[off[e] + pos] = t;
        entryW[off[e] + pos]   = topw[g];
    }
}

// ---------------------------------------------------------------------------
extern "C" void kernel_launch(void* const* d_in, const int* in_sizes, int n_in,
                              void* d_out, int out_size, void* d_ws, size_t ws_size,
                              hipStream_t stream)
{
    (void)in_sizes; (void)n_in; (void)out_size; (void)ws_size;
    const float* x    = (const float*)d_in[0];   // hidden_states (1,1024,2048)
    const float* pe   = (const float*)d_in[1];   // position_embeddings (1024,64)
    // d_in[2] attention_mask (causal, applied analytically); d_in[3] start_pos (0)
    const float* ln1w = (const float*)d_in[4];
    const float* ln2w = (const float*)d_in[5];
    const float* q_w  = (const float*)d_in[6];   // (2048,4096)
    const float* k_w  = (const float*)d_in[7];   // (2048,512)
    const float* v_w  = (const float*)d_in[8];   // (2048,512)
    const float* o_w  = (const float*)d_in[9];   // (4096,2048)
    const float* qn_w = (const float*)d_in[10];  // (128,)
    const float* kn_w = (const float*)d_in[11];  // (128,)
    const float* gw   = (const float*)d_in[12];  // (2048,32)
    const float* wg   = (const float*)d_in[13];  // (32,2048,768)
    const float* wu   = (const float*)d_in[14];  // (32,2048,768)
    const float* wd   = (const float*)d_in[15];  // (32,768,2048)
    float* out = (float*)d_out;                  // (1,1024,2048)

    // workspace layout
    float* ws    = (float*)d_ws;
    float* h1    = ws;                          // 1024*2048
    float* qb    = h1 + Sq * Hd;                // 1024*4096 (pre-rope fp32 q)
    float* kb    = qb + Sq * NHq * HDq;         // 1024*512
    float* vb    = kb + Sq * NKV * HDq;         // 1024*512
    float* scAll = vb + Sq * NKV * HDq;         // 8*1024*1024 fp32 scores
    float* ob    = scAll + (size_t)8 * Sq * Sq; // 1024*4096
    float* h2    = ob + Sq * NHq * HDq;         // 1024*2048
    float* topw  = h2 + Sq * Hd;                // 1024*8
    float* entW  = topw + Sq * TOPK;            // 1024*8
    float* act   = entW + Sq * TOPK;            // 1024*8*768
    _Float16* qh = (_Float16*)(act + (size_t)Sq * TOPK * NI);   // 1024*4096 f16
    _Float16* kh = qh + (size_t)Sq * NHq * HDq;                 // 1024*512 f16
    _Float16* vh = kh + (size_t)Sq * NKV * HDq;                 // 1024*512 f16
    _Float16* pf = vh + (size_t)Sq * NKV * HDq;                 // 8*1024*1024 f16 probs
    int* ip      = (int*)(pf + (size_t)8 * Sq * Sq);
    int* topi    = ip;                          // 1024*8
    int* cnt     = topi + Sq * TOPK;            // 32
    int* cnt2    = cnt + NE;                    // 32
    int* offp    = cnt2 + NE;                   // 32
    int* entTok  = offp + NE;                   // 1024*8

    const dim3 blk256(256);

    // 1) ln1
    rmsnorm_kernel<<<Sq, blk256, 0, stream>>>(x, ln1w, h1);

    // 2) QKV projections (128x128 weight GEMM, fp32 -> fp32)
    gemm128_f16w<false><<<dim3(NHq * HDq / 128, Sq / 128), blk256, 0, stream>>>(
        h1, q_w, nullptr, qb, Sq, NHq * HDq, Hd, Hd, NHq * HDq, NHq * HDq);
    gemm128_f16w<false><<<dim3(NKV * HDq / 128, Sq / 128), blk256, 0, stream>>>(
        h1, k_w, nullptr, kb, Sq, NKV * HDq, Hd, Hd, NKV * HDq, NKV * HDq);
    gemm128_f16w<false><<<dim3(NKV * HDq / 128, Sq / 128), blk256, 0, stream>>>(
        h1, v_w, nullptr, vb, Sq, NKV * HDq, Hd, Hd, NKV * HDq, NKV * HDq);

    // 3) q/k head RMSNorm + RoPE -> f16 side buffers; cast V to f16
    qknorm_rope_kernel<<<dim3(Sq, NHq), 64, 0, stream>>>(qb, qh, qn_w, pe, NHq);
    qknorm_rope_kernel<<<dim3(Sq, NKV), 64, 0, stream>>>(kb, kh, kn_w, pe, NKV);
    cast_f16_kernel<<<(Sq * NKV * HDq / 4 + 255) / 256, blk256, 0, stream>>>(
        vb, vh, Sq * NKV * HDq);

    // 4) attention, one KV-group (8 heads) per step, heads batched on grid.z
    for (int kv = 0; kv < NKV; ++kv) {
        const int hBase = kv * (NHq / NKV);
        // scores[z] = Q_{hBase+z} @ K_kv^T   (f16 operands)
        gemm_tile64<_Float16, _Float16, true, false><<<dim3(Sq / 64, Sq / 64, 8), blk256, 0, stream>>>(
            qh + hBase * HDq, kh + kv * HDq, nullptr, scAll,
            Sq, Sq, HDq, NHq * HDq, NKV * HDq, Sq,
            (long long)HDq, 0LL, (long long)Sq * Sq);
        attn_softmax_kernel<<<dim3(Sq, 8), blk256, 0, stream>>>(scAll, pf);
        // O_{hBase+z} = P[z] @ V_kv          (f16 operands)
        gemm_tile64<_Float16, _Float16, false, false><<<dim3(HDq / 64, Sq / 64, 8), blk256, 0, stream>>>(
            pf, vh + kv * HDq, nullptr, ob + hBase * HDq,
            Sq, HDq, Sq, Sq, NKV * HDq, NHq * HDq,
            (long long)Sq * Sq, 0LL, (long long)HDq);
    }

    // 5) output projection with fused residual: out = x + ob @ o_w
    gemm128_f16w<true><<<dim3(Hd / 128, Sq / 128), blk256, 0, stream>>>(
        ob, o_w, x, out, Sq, Hd, NHq * HDq, NHq * HDq, Hd, Hd);

    // 6) ln2
    rmsnorm_kernel<<<Sq, blk256, 0, stream>>>(out, ln2w, h2);

    // 7) router + per-expert token lists
    gate_topk_kernel<<<Sq, 32, 0, stream>>>(h2, gw, topw, topi);
    zero_counts_kernel<<<1, 64, 0, stream>>>(cnt, cnt2);
    count_kernel<<<(Sq * TOPK + 255) / 256, blk256, 0, stream>>>(topi, cnt);
    scan_kernel<<<1, 1, 0, stream>>>(cnt, offp);
    scatter_kernel<<<(Sq * TOPK + 255) / 256, blk256, 0, stream>>>(topi, topw, offp, cnt2, entTok, entW);

    // 8) expert up (fused gate/up/SiLU) and down (weighted atomic scatter into out)
    moe_up_kernel<<<dim3(NI / 64, Sq / 128, NE), blk256, 0, stream>>>(
        h2, wg, wu, act, cnt, offp, entTok);
    moe_down_kernel<<<dim3(Hd / 128, Sq / 128, NE), blk256, 0, stream>>>(
        act, wd, out, cnt, offp, entTok, entW);
}
